// Qwen3OmniMoeMLP_84404697301641
// MI455X (gfx1250) — matrix operand order repaired and checked by reading, not compiled, not run on hardware
//
#include <hip/hip_runtime.h>

// ---------------------------------------------------------------------------
// MoE MLP (Qwen3-style) for MI455X / gfx1250.
//   H=1024, I=2048, E=16, top-2, nt=4096 tokens, fp32 tensors.
// Memory floor: W1+W2 = 256MB fp32 @ 23.3TB/s ~ 11us.  Routed compute:
// ~69 GFLOP on v_wmma_f32_16x16x32_bf16 (fp32->bf16 via v_cvt_pk_bf16_f32).
// TILE_M=32: each weight B-fragment feeds 2 WMMAs; A-fragments are loaded
// as 2x ds_load_b128; x staging stores b128.
// ---------------------------------------------------------------------------

#define HID   1024
#define IDIM  2048
#define NEXP  16
#define NTOK  4096
#define TILE_M 32
#define ICHUNK 512
#define MAX_TILES (NTOK / TILE_M)   // 128 worst-case tiles per expert

typedef __attribute__((ext_vector_type(16))) __bf16 v16bf;
typedef __attribute__((ext_vector_type(8)))  float  v8f;
typedef __attribute__((ext_vector_type(2)))  float  v2f;
typedef __attribute__((ext_vector_type(2)))  __bf16 v2bf;

union FragU { unsigned int u[8]; uint4 q[2]; v16bf bf; };

// native packed f32->bf16 (RNE): one v_cvt_pk_bf16_f32
__device__ __forceinline__ unsigned int pack2(float a, float b) {
  v2f f = {a, b};
  v2bf h = __builtin_convertvector(f, v2bf);
  return __builtin_bit_cast(unsigned int, h);
}

__device__ __forceinline__ float silu(float v) {
  return v * __builtin_amdgcn_rcpf(1.f + __expf(-v));
}

// A-fragment (16-bit A 16x32 layout): two 16B LDS loads.
// For lane half h: r=0..7 hold K = 16h+{0,2,...,14} (8 consecutive dwords).
__device__ __forceinline__ FragU loadA(const unsigned int* base_dw,
                                       int row, int ld, int kc, int half) {
  FragU A;
  int d0 = (row * ld + kc + 16 * half) >> 1;
  A.q[0] = *(const uint4*)(base_dw + d0);
  A.q[1] = *(const uint4*)(base_dw + d0 + 4);
  return A;
}

// ---------------------------------------------------------------------------
// Gating: one wave32 per token; softmax + top-2 via shuffles; lane 0 appends
// (token, prob) to the two chosen experts' lists.
// ---------------------------------------------------------------------------
__global__ __launch_bounds__(32) void gate_kernel(
    const float* __restrict__ x, const float* __restrict__ Wg,
    int* __restrict__ counts, int* __restrict__ toks, float* __restrict__ wts) {
  int t = blockIdx.x;
  int lane = threadIdx.x;
  int e = lane & 15;
  int half = lane >> 4;

  const float* xr = x + (size_t)t * HID + half * (HID / 2);
  const float* wr = Wg + (size_t)e * HID + half * (HID / 2);
  float acc = 0.f;
  for (int k = 0; k < HID / 2; k += 4) {
    float4 xv = *(const float4*)(xr + k);
    float4 wv = *(const float4*)(wr + k);
    acc += xv.x * wv.x + xv.y * wv.y + xv.z * wv.z + xv.w * wv.w;
  }
  float logit = acc + __shfl_xor(acc, 16, 32);

  float mx = logit;
  for (int m = 1; m <= 8; m <<= 1) mx = fmaxf(mx, __shfl_xor(mx, m, 32));
  float ex = __expf(logit - mx);
  float s = ex;
  for (int m = 1; m <= 8; m <<= 1) s += __shfl_xor(s, m, 32);
  float p = ex / s;

  float p1 = (half == 0) ? p : -1.f;
  int   i1 = e;
  for (int m = 1; m <= 8; m <<= 1) {
    float op = __shfl_xor(p1, m, 32);
    int   oi = __shfl_xor(i1, m, 32);
    if (op > p1 || (op == p1 && oi < i1)) { p1 = op; i1 = oi; }
  }
  float p2 = (half == 0 && e != i1) ? p : -1.f;
  int   i2 = e;
  for (int m = 1; m <= 8; m <<= 1) {
    float op = __shfl_xor(p2, m, 32);
    int   oi = __shfl_xor(i2, m, 32);
    if (op > p2 || (op == p2 && oi < i2)) { p2 = op; i2 = oi; }
  }

  if (lane == 0) {
    int s1 = atomicAdd(&counts[i1], 1);
    toks[i1 * NTOK + s1] = t;  wts[i1 * NTOK + s1] = p1;
    int s2 = atomicAdd(&counts[i2], 1);
    toks[i2 * NTOK + s2] = t;  wts[i2 * NTOK + s2] = p2;
  }
}

// ---------------------------------------------------------------------------
// Expert MLP: block = (expert e, tile of 32 routed tokens), 8 waves.
//   phase 0: gather x rows -> bf16 LDS tile [32 x 1024]           (64 KB)
//   loop I in chunks of 512:
//     phase 1: h = silu(x @ W1e^T) chunk -> bf16 LDS [32 x 512]   (32 KB)
//     phase 2: y += h_chunk @ W2e^T chunk  (f32 accum: 2x8 v8f per wave)
//   epilogue: y *= combine weight, global_atomic_add_f32 scatter.
// ---------------------------------------------------------------------------
__global__ __launch_bounds__(256) void moe_kernel(
    const float* __restrict__ x,  const float* __restrict__ W1,
    const float* __restrict__ W2, const int* __restrict__ counts,
    const int* __restrict__ toks, const float* __restrict__ wts,
    float* __restrict__ out) {
  int e    = blockIdx.x / MAX_TILES;
  int tile = blockIdx.x % MAX_TILES;
  int cnt  = counts[e];
  int m0   = tile * TILE_M;
  if (m0 >= cnt) return;                 // uniform early exit

  __shared__ unsigned int xs[TILE_M * HID / 2];     // bf16 x tile (64 KB)
  __shared__ unsigned int hs[TILE_M * ICHUNK / 2];  // bf16 h chunk (32 KB)
  __shared__ int   rows[TILE_M];
  __shared__ float wrow[TILE_M];

  int tid  = threadIdx.x;
  int lane = tid & 31;
  int wave = tid >> 5;
  int half = lane >> 4;
  int lm   = lane & 15;

  if (tid < TILE_M) {
    int g  = m0 + tid;
    int gc = g < cnt ? g : cnt - 1;      // pad with a valid row
    rows[tid] = toks[e * NTOK + gc];
    wrow[tid] = (g < cnt) ? wts[e * NTOK + g] : 0.f;
  }
  __syncthreads();

  // phase 0: stage gathered x rows as bf16, 8 floats -> one b128 store
  for (int j = tid * 4; j < TILE_M * HID / 2; j += 256 * 4) {
    int row = j / (HID / 2);
    int kp  = (j % (HID / 2)) * 2;
    const float* src = x + (size_t)rows[row] * HID + kp;
    float4 f0 = *(const float4*)(src);
    float4 f1 = *(const float4*)(src + 4);
    uint4 d;
    d.x = pack2(f0.x, f0.y);  d.y = pack2(f0.z, f0.w);
    d.z = pack2(f1.x, f1.y);  d.w = pack2(f1.z, f1.w);
    *(uint4*)&xs[j] = d;
  }

  v8f acc2[2][8] = {};                   // y tile: 2 M-subtiles x 8 N-tiles
  __syncthreads();

  const float* W1e = W1 + (size_t)e * IDIM * HID;
  const float* W2e = W2 + (size_t)e * HID * IDIM;

  for (int ic0 = 0; ic0 < IDIM; ic0 += ICHUNK) {
    // ---- phase 1: h chunk = silu(x @ W1e^T) ----
    v8f acc1[2][4] = {};
    for (int kc = 0; kc < HID; kc += 32) {
      FragU A0 = loadA(xs, lm,      HID, kc, half);
      FragU A1 = loadA(xs, lm + 16, HID, kc, half);
#pragma unroll
      for (int t = 0; t < 4; ++t) {
        int i0 = ic0 + (wave * 4 + t) * 16;
        const float* bp = W1e + (size_t)(i0 + lm) * HID + kc + half * 16;
        __builtin_prefetch(bp + 32, 0, 3);
        FragU B;                         // 16 consecutive K per lane
#pragma unroll
        for (int q = 0; q < 4; ++q) {
          float4 f = *(const float4*)(bp + q * 4);
          B.u[q * 2]     = pack2(f.x, f.y);
          B.u[q * 2 + 1] = pack2(f.z, f.w);
        }
        acc1[0][t] = __builtin_amdgcn_wmma_f32_16x16x32_bf16(
            false, A0.bf, false, B.bf, (short)0, acc1[0][t], false, false);
        acc1[1][t] = __builtin_amdgcn_wmma_f32_16x16x32_bf16(
            false, A1.bf, false, B.bf, (short)0, acc1[1][t], false, false);
      }
    }
    // silu + store h chunk (D layout: M = r + 8*half, N = lm)
#pragma unroll
    for (int t = 0; t < 4; ++t) {
      int iloc = (wave * 4 + t) * 16 + lm;
#pragma unroll
      for (int mt = 0; mt < 2; ++mt) {
#pragma unroll
        for (int r = 0; r < 8; ++r) {
          int m = mt * 16 + r + 8 * half;
          ((__bf16*)hs)[m * ICHUNK + iloc] = (__bf16)silu(acc1[mt][t][r]);
        }
      }
    }
    __syncthreads();

    // ---- phase 2: y += h_chunk @ W2e^T ----
    for (int kc = 0; kc < ICHUNK; kc += 32) {
      FragU A0 = loadA(hs, lm,      ICHUNK, kc, half);
      FragU A1 = loadA(hs, lm + 16, ICHUNK, kc, half);
#pragma unroll
      for (int t = 0; t < 8; ++t) {
        int n0 = wave * 128 + t * 16;
        const float* bp = W2e + (size_t)(n0 + lm) * IDIM + ic0 + kc + half * 16;
        __builtin_prefetch(bp + 32, 0, 3);
        FragU B;
#pragma unroll
        for (int q = 0; q < 4; ++q) {
          float4 f = *(const float4*)(bp + q * 4);
          B.u[q * 2]     = pack2(f.x, f.y);
          B.u[q * 2 + 1] = pack2(f.z, f.w);
        }
        acc2[0][t] = __builtin_amdgcn_wmma_f32_16x16x32_bf16(
            false, A0.bf, false, B.bf, (short)0, acc2[0][t], false, false);
        acc2[1][t] = __builtin_amdgcn_wmma_f32_16x16x32_bf16(
            false, A1.bf, false, B.bf, (short)0, acc2[1][t], false, false);
      }
    }
    __syncthreads();
  }

  // epilogue: scale by combine weight, scatter-accumulate
#pragma unroll
  for (int t = 0; t < 8; ++t) {
    int n = wave * 128 + t * 16 + lm;
#pragma unroll
    for (int mt = 0; mt < 2; ++mt) {
#pragma unroll
      for (int r = 0; r < 8; ++r) {
        int m = mt * 16 + r + 8 * half;
        if (m0 + m < cnt) {
          atomicAdd(&out[(size_t)rows[m] * HID + n], wrow[m] * acc2[mt][t][r]);
        }
      }
    }
  }
}

// ---------------------------------------------------------------------------
extern "C" void kernel_launch(void* const* d_in, const int* in_sizes, int n_in,
                              void* d_out, int out_size, void* d_ws, size_t ws_size,
                              hipStream_t stream) {
  const float* x  = (const float*)d_in[0];   // [4096, 1024]
  const float* Wg = (const float*)d_in[1];   // [16, 1024]
  const float* W1 = (const float*)d_in[2];   // [16, 2048, 1024]
  const float* W2 = (const float*)d_in[3];   // [16, 1024, 2048]
  float* out = (float*)d_out;                // [4096, 1024]

  int*   counts = (int*)d_ws;
  int*   toks   = counts + 64;
  float* wts    = (float*)(toks + NEXP * NTOK);

  hipMemsetAsync(counts, 0, 64 * sizeof(int), stream);
  hipMemsetAsync(out, 0, (size_t)out_size * sizeof(float), stream);

  gate_kernel<<<NTOK, 32, 0, stream>>>(x, Wg, counts, toks, wts);
  moe_kernel<<<NEXP * MAX_TILES, 256, 0, stream>>>(x, W1, W2, counts, toks, wts, out);
}
